// QuestionSelfAttention_77446850282041
// MI455X (gfx1250) — compile-verified, matrix operand order checked by reading, not run on hardware
//
#include <hip/hip_runtime.h>
#include <hip/hip_bf16.h>
#include <math.h>

// ---------------------------------------------------------------------------
// QuestionSelfAttention for MI455X (gfx1250), wave32 + WMMA.
//
//   h      = tanh(q @ W1 + b1)              [B*S, H]   (never materialized)
//   logit  = h @ W2 (+ b2, softmax-invariant -> dropped)
//   attnT  = softmax over batch axis of logit^T  -> stored [S, B] flat
//   out[i] = sum_j attnT_flat[i*S + j] * q[i, j, :]
//
// MI455X-specific design points:
//  - v_wmma_f32_16x16x32_bf16 (fp32 accum) for the 481-GFLOP GEMM
//  - full-K W1 column slice (1024x64 bf16 = 130KB) resident in the WGP's
//    320KB LDS, staged once per block; 14 M-tiles streamed against it
//  - double-buffered A tile: one s_barrier per K-step, global->reg prefetch
//    overlapped with WMMA
//  - branchless tanh (v_exp_f32 + v_rcp_f32) in the fused epilogue
// ---------------------------------------------------------------------------

typedef __attribute__((ext_vector_type(16))) __bf16 v16bf;
typedef __attribute__((ext_vector_type(8)))  __bf16 bf16x8;
typedef __attribute__((ext_vector_type(4)))  __bf16 bf16x4;
typedef __attribute__((ext_vector_type(8)))  float  v8f;

#define B_DIM 16384
#define S_DIM 14
#define H_DIM 1024
#define M_DIM (B_DIM * S_DIM)   // 229376 rows

#define BM 128
#define BN 64
#define BK 32
#define NK (H_DIM / BK)          // 32 K-steps
#define TILES_PER_BLOCK 14       // 1792 M-tiles / 128 blocks
#define A_STRIDE 40              // elements; 80B rows, 16B chunk alignment
#define W_STRIDE 1040            // elements; full-K row, multiple of 8

// Branchless tanh: 1 - 2/(exp(2x)+1). Saturates to +/-1 for large |x|
// (exp -> inf or 0), no exec-mask branching; ~5 VALU ops on gfx1250.
__device__ __forceinline__ float fast_tanhf(float x) {
    const float e = __expf(2.0f * x);
    return 1.0f - 2.0f * __builtin_amdgcn_rcpf(e + 1.0f);
}

__global__ void qsa_gemm_logits_kernel(const float* __restrict__ q,
                                       const float* __restrict__ W1,
                                       const float* __restrict__ b1,
                                       const float* __restrict__ W2,
                                       float* __restrict__ logitsT) {
    // 150KB static LDS: legal on gfx1250 (320KB/WGP), 2 workgroups per WGP.
    __shared__ __align__(16) __bf16 sA[2][BM][A_STRIDE];    //  20.0 KB
    __shared__ __align__(16) __bf16 sW[BN][W_STRIDE];       // 130.0 KB, [n][k]

    const int n0 = blockIdx.y * BN;

    const int tid  = threadIdx.x;
    const int lane = tid & 31;
    const int wave = tid >> 5;
    const int wr   = wave >> 1;      // 0..3 -> 32-row band
    const int wc   = wave & 1;       // 0..1 -> 32-col band
    const int half = lane >> 4;      // selects K-halves / row+8 in C layout
    const int l16  = lane & 15;

    // loader lane mappings
    const int tA_row = tid >> 3;           // 0..31 (8 threads x float4 per row)
    const int tA_col = (tid & 7) * 4;
    const int tW_row = tid >> 4;           // 0..15 (16 threads x float4 per row)
    const int tW_col = (tid & 15) * 4;

    // ---- Phase 1: stage the full-K W1 slice [1024][n0..n0+63] -> sW[n][k] ----
    #pragma unroll 4
    for (int pass = 0; pass < 64; ++pass) {
        const int kr = pass * 16 + tW_row;
        const float4 v = *(const float4*)&W1[(size_t)kr * H_DIM + n0 + tW_col];
        sW[tW_col + 0][kr] = (__bf16)v.x;
        sW[tW_col + 1][kr] = (__bf16)v.y;
        sW[tW_col + 2][kr] = (__bf16)v.z;
        sW[tW_col + 3][kr] = (__bf16)v.w;
    }

    // epilogue constants (same n-columns for every M-tile of this block)
    float w2v[2], b1v[2];
    #pragma unroll
    for (int tj = 0; tj < 2; ++tj) {
        const int n = n0 + wc * 32 + tj * 16 + l16;
        w2v[tj] = W2[n];
        b1v[tj] = b1[n];
    }
    __syncthreads();

    // ---- Phase 2: stream 14 M-tiles against the resident W1 slice ----
    for (int t = 0; t < TILES_PER_BLOCK; ++t) {
        const int m0 = (blockIdx.x * TILES_PER_BLOCK + t) * BM;
        const float* qbase = q + (size_t)(m0 + tA_row) * H_DIM + tA_col;

        v8f acc[2][2] = {};
        float4 r[4];

        // prologue: fill buffer 0
        #pragma unroll
        for (int p = 0; p < 4; ++p)
            r[p] = *(const float4*)(qbase + (size_t)p * 32 * H_DIM);
        #pragma unroll
        for (int p = 0; p < 4; ++p) {
            bf16x4 pk;
            pk[0] = (__bf16)r[p].x; pk[1] = (__bf16)r[p].y;
            pk[2] = (__bf16)r[p].z; pk[3] = (__bf16)r[p].w;
            *(bf16x4*)&sA[0][p * 32 + tA_row][tA_col] = pk;
        }
        __syncthreads();

        for (int kt = 0; kt < NK; ++kt) {
            const int cur = kt & 1;
            const int k0  = kt * BK;

            // overlap: fetch next K tile of q into registers
            if (kt + 1 < NK) {
                #pragma unroll
                for (int p = 0; p < 4; ++p)
                    r[p] = *(const float4*)(qbase + (size_t)p * 32 * H_DIM + (kt + 1) * BK);
                // gfx1250 global_prefetch for the K tile after that
                if (kt + 2 < NK)
                    __builtin_prefetch(qbase + (kt + 2) * BK, 0, 1);
            }

            // fragments per the 16-bit A 16x32 layout:
            // lane l, element e: K = (e/8)*16 + (l/16)*8 + (e%8)
            v16bf afrag[2], bfrag[2];
            #pragma unroll
            for (int ti = 0; ti < 2; ++ti) {
                const int row = wr * 32 + ti * 16 + l16;
                const bf16x8 lo = *(const bf16x8*)&sA[cur][row][half * 8];
                const bf16x8 hi = *(const bf16x8*)&sA[cur][row][16 + half * 8];
                #pragma unroll
                for (int e = 0; e < 8; ++e) { afrag[ti][e] = lo[e]; afrag[ti][8 + e] = hi[e]; }
            }
            #pragma unroll
            for (int tj = 0; tj < 2; ++tj) {
                const int col = wc * 32 + tj * 16 + l16;
                const bf16x8 lo = *(const bf16x8*)&sW[col][k0 + half * 8];
                const bf16x8 hi = *(const bf16x8*)&sW[col][k0 + 16 + half * 8];
                #pragma unroll
                for (int e = 0; e < 8; ++e) { bfrag[tj][e] = lo[e]; bfrag[tj][8 + e] = hi[e]; }
            }

            #pragma unroll
            for (int ti = 0; ti < 2; ++ti)
                #pragma unroll
                for (int tj = 0; tj < 2; ++tj)
                    acc[ti][tj] = __builtin_amdgcn_wmma_f32_16x16x32_bf16(
                        false, afrag[ti], false, bfrag[tj],
                        (short)0, acc[ti][tj], false, false);

            // store next tile into the other buffer (no race: cur != nxt)
            if (kt + 1 < NK) {
                #pragma unroll
                for (int p = 0; p < 4; ++p) {
                    bf16x4 pk;
                    pk[0] = (__bf16)r[p].x; pk[1] = (__bf16)r[p].y;
                    pk[2] = (__bf16)r[p].z; pk[3] = (__bf16)r[p].w;
                    *(bf16x4*)&sA[cur ^ 1][p * 32 + tA_row][tA_col] = pk;
                }
            }
            __syncthreads();
        }

        // ---- fused epilogue: tanh, dot with W2, row-reduce, scatter [S,B] ----
        // C layout: VGPR r holds row (ti*16 + half*8 + r), lane%16 is the col.
        #pragma unroll
        for (int ti = 0; ti < 2; ++ti) {
            #pragma unroll
            for (int rr = 0; rr < 8; ++rr) {
                float v = fast_tanhf(acc[ti][0][rr] + b1v[0]) * w2v[0]
                        + fast_tanhf(acc[ti][1][rr] + b1v[1]) * w2v[1];
                v += __shfl_xor(v, 1, 32);
                v += __shfl_xor(v, 2, 32);
                v += __shfl_xor(v, 4, 32);
                v += __shfl_xor(v, 8, 32);
                if (l16 == 0) {
                    const int m = m0 + wr * 32 + ti * 16 + half * 8 + rr;
                    const int b = m / S_DIM;
                    const int s = m % S_DIM;
                    atomicAdd(&logitsT[(size_t)s * B_DIM + b], v);
                }
            }
        }
    }
}

// Softmax over the batch axis of logitsT[S, B], in place (each thread reads
// its 16 values into registers before any write). 14 blocks x 1024 threads.
__global__ void qsa_softmax_kernel(float* __restrict__ attnT) {
    const int s = blockIdx.x;
    const int t = threadIdx.x;
    float* row = attnT + (size_t)s * B_DIM;

    float vals[16];
    float m = -INFINITY;
    #pragma unroll
    for (int i = 0; i < 16; ++i) {
        vals[i] = row[t + i * 1024];
        m = fmaxf(m, vals[i]);
    }
    __shared__ float red[1024];
    red[t] = m;
    __syncthreads();
    for (int off = 512; off > 0; off >>= 1) {
        if (t < off) red[t] = fmaxf(red[t], red[t + off]);
        __syncthreads();
    }
    const float gmax = red[0];
    __syncthreads();

    float e[16];
    float sum = 0.0f;
    #pragma unroll
    for (int i = 0; i < 16; ++i) {
        e[i] = __expf(vals[i] - gmax);
        sum += e[i];
    }
    red[t] = sum;
    __syncthreads();
    for (int off = 512; off > 0; off >>= 1) {
        if (t < off) red[t] += red[t + off];
        __syncthreads();
    }
    const float inv = 1.0f / red[0];
    #pragma unroll
    for (int i = 0; i < 16; ++i)
        row[t + i * 1024] = e[i] * inv;
}

// Weighted pooling: out[i, :] = sum_j attnT_flat[i*S + j] * q[i, j, :].
// Memory-bound; float4 everywhere. One block per batch row i.
__global__ void qsa_pool_kernel(const float* __restrict__ q,
                                const float* __restrict__ attn_flat,
                                float* __restrict__ out) {
    const int i = blockIdx.x;
    const int t = threadIdx.x;        // 256 threads, 4 floats each
    __shared__ float w[S_DIM];
    if (t < S_DIM) w[t] = attn_flat[(size_t)i * S_DIM + t];
    __syncthreads();

    const int h = t * 4;
    float4 acc = {0.f, 0.f, 0.f, 0.f};
    #pragma unroll
    for (int j = 0; j < S_DIM; ++j) {
        const float4 qv = *(const float4*)&q[((size_t)i * S_DIM + j) * H_DIM + h];
        const float wj = w[j];
        acc.x += wj * qv.x;
        acc.y += wj * qv.y;
        acc.z += wj * qv.z;
        acc.w += wj * qv.w;
    }
    *(float4*)&out[(size_t)i * H_DIM + h] = acc;
}

extern "C" void kernel_launch(void* const* d_in, const int* in_sizes, int n_in,
                              void* d_out, int out_size, void* d_ws, size_t ws_size,
                              hipStream_t stream) {
    const float* q  = (const float*)d_in[0];   // [B, S, H]
    const float* W1 = (const float*)d_in[1];   // [H, H]
    const float* b1 = (const float*)d_in[2];   // [H]
    const float* W2 = (const float*)d_in[3];   // [H, 1]
    // d_in[4] = b2: softmax-invariant, intentionally unused
    (void)in_sizes; (void)n_in; (void)out_size; (void)ws_size;

    float* out     = (float*)d_out;            // [B, H]
    float* logitsT = (float*)d_ws;             // [S, B] = attn weights in place

    hipMemsetAsync(logitsT, 0, (size_t)M_DIM * sizeof(float), stream);

    dim3 gGemm(M_DIM / (BM * TILES_PER_BLOCK), H_DIM / BN);   // 128 x 16
    qsa_gemm_logits_kernel<<<gGemm, 256, 0, stream>>>(q, W1, b1, W2, logitsT);

    qsa_softmax_kernel<<<S_DIM, 1024, 0, stream>>>(logitsT);

    qsa_pool_kernel<<<B_DIM, 256, 0, stream>>>(q, logitsT, out);
}
